// HybridSeqModelingBlock_36086315221749
// MI455X (gfx1250) — matrix operand order repaired and checked by reading, not compiled, not run on hardware
//
#include <hip/hip_runtime.h>
#include <cstdint>
#include <cstddef>

// ---------------------------------------------------------------------------
// CDNA5 (gfx1250) implementation of the hybrid attention + TTT block.
// Matrix math: v_wmma_f32_16x16x32_bf16 (wave32 WMMA).
// Data movement: global_load_async_to_lds_b128 (ASYNCcnt) for tile staging,
// fragment-major ("packed") B operand storage in LDS so every WMMA operand
// load is a contiguous ds_load_b128 pair (no per-element column gathers).
// ---------------------------------------------------------------------------

typedef __attribute__((ext_vector_type(16))) __bf16 v16bf;
typedef __attribute__((ext_vector_type(8)))  __bf16 v8bf;
typedef __attribute__((ext_vector_type(8)))  float  v8f;
typedef __attribute__((ext_vector_type(4)))  float  v4f;
typedef __attribute__((ext_vector_type(4)))  int    v4i;

constexpr int kB   = 4;
constexpr int kS   = 1024;
constexpr int kD   = 1024;
constexpr int kH   = 16;
constexpr int kHD  = 64;
constexpr int kC   = 64;
constexpr int kNC  = 16;
constexpr int kDFF = 256;
constexpr float kEPS = 1e-6f;

// ----------------------- async global->LDS (gfx1250) ------------------------

#if defined(__AMDGCN__) && __has_builtin(__builtin_amdgcn_global_load_async_to_lds_b128)
#define USE_ASYNC_LDS 1
#else
#define USE_ASYNC_LDS 0
#endif

// Copy 16 bytes global -> LDS. Async (ASYNCcnt-tracked) when available.
__device__ __forceinline__ void async_copy16(const void* gsrc, void* ldst) {
#if USE_ASYNC_LDS
  __builtin_amdgcn_global_load_async_to_lds_b128(
      (__attribute__((address_space(1))) v4i*)gsrc,
      (__attribute__((address_space(3))) v4i*)ldst, 0, 0);
#else
  *(uint4*)ldst = *(const uint4*)gsrc;
#endif
}

__device__ __forceinline__ void wait_async_lds() {
#if USE_ASYNC_LDS
#if __has_builtin(__builtin_amdgcn_s_wait_asynccnt)
  __builtin_amdgcn_s_wait_asynccnt(0);
#else
  asm volatile("s_wait_asynccnt 0" ::: "memory");
#endif
#endif
}

// ------------------------------ WMMA helpers --------------------------------

__device__ __forceinline__ v8f wmma_bf16(v16bf a, v16bf b, v8f c) {
  return __builtin_amdgcn_wmma_f32_16x16x32_bf16(false, a, false, b, (short)0, c, false, false);
}

// A fragment: 16(M) x 32(K) bf16, row-major source, lda in elements.
// Lane l: m = l&15, g = l>>4; element j holds k = 16g + j (contiguous).
__device__ __forceinline__ v16bf load_a_frag(const __bf16* A, int lda) {
  int l = threadIdx.x & 31, m = l & 15, g = l >> 4;
  const __bf16* base = A + m * lda + 16 * g;
  v8bf c0 = *(const v8bf*)(base);
  v8bf c1 = *(const v8bf*)(base + 8);
  v16bf a;
#pragma unroll
  for (int j = 0; j < 8; ++j) { a[j] = c0[j]; a[j + 8] = c1[j]; }
  return a;
}

// Same, from f32 source (converts).
__device__ __forceinline__ v16bf load_a_frag_f32(const float* A, int lda) {
  int l = threadIdx.x & 31, m = l & 15, g = l >> 4;
  const float* base = A + m * lda + 16 * g;
  v4f c0 = *(const v4f*)(base);
  v4f c1 = *(const v4f*)(base + 4);
  v4f c2 = *(const v4f*)(base + 8);
  v4f c3 = *(const v4f*)(base + 12);
  v16bf a;
#pragma unroll
  for (int j = 0; j < 4; ++j) {
    a[j]      = (__bf16)c0[j];
    a[j + 4]  = (__bf16)c1[j];
    a[j + 8]  = (__bf16)c2[j];
    a[j + 12] = (__bf16)c3[j];
  }
  return a;
}

// B fragment: 32(K) x 16(N) from row-major [K][ldb] source (column gather).
__device__ __forceinline__ v16bf load_b_frag(const __bf16* Bm, int ldb) {
  int l = threadIdx.x & 31, n = l & 15, g = l >> 4;
  v16bf b;
#pragma unroll
  for (int j = 0; j < 16; ++j) b[j] = Bm[(16 * g + j) * ldb + n];
  return b;
}

// B fragment from a transposed (N-major) source: contiguous per lane.
__device__ __forceinline__ v16bf load_b_frag_T(const __bf16* Bm, int ldb) {
  int l = threadIdx.x & 31, n = l & 15, g = l >> 4;
  const __bf16* base = Bm + n * ldb + 16 * g;
  v8bf c0 = *(const v8bf*)(base);
  v8bf c1 = *(const v8bf*)(base + 8);
  v16bf b;
#pragma unroll
  for (int j = 0; j < 8; ++j) { b[j] = c0[j]; b[j + 8] = c1[j]; }
  return b;
}

// ---- fragment-major ("packed") B storage ----
// A K x N bf16 matrix is stored as tiles (kt = k/32, nt = n/16); within a tile,
// slot layout is [lane][j] where lane = (n&15) + 16*((k>>4)&1), j = k&15.
// Fragment loads then become 32 contiguous bytes per lane.
__device__ __forceinline__ int bpack(int k, int n, int ntiles) {
  return ((((k >> 5) * ntiles) + (n >> 4)) * 32 + ((n & 15) + ((k >> 4) & 1) * 16)) * 16 + (k & 15);
}

__device__ __forceinline__ v16bf load_b_frag_packed(const __bf16* Bp, int kt, int nt, int ntiles) {
  int l = threadIdx.x & 31;
  const __bf16* base = Bp + ((kt * ntiles + nt) * 32 + l) * 16;
  v8bf c0 = *(const v8bf*)(base);
  v8bf c1 = *(const v8bf*)(base + 8);
  v16bf b;
#pragma unroll
  for (int j = 0; j < 8; ++j) { b[j] = c0[j]; b[j + 8] = c1[j]; }
  return b;
}

// C/D fragment: lane l: n = l&15, g = l>>4; vgpr r -> row M = 8g + r
__device__ __forceinline__ void store_c_frag(float* Cf, int ldc, v8f c) {
  int l = threadIdx.x & 31, n = l & 15, g = l >> 4;
#pragma unroll
  for (int r = 0; r < 8; ++r) Cf[(8 * g + r) * ldc + n] = c[r];
}

__device__ __forceinline__ float wave_sum(float v) {
#pragma unroll
  for (int off = 16; off; off >>= 1) v += __shfl_xor(v, off, 32);
  return v;
}

__device__ __forceinline__ float gelu_f(float x) {
  float t = tanhf(0.79788456f * (x + 0.044715f * x * x * x));
  return 0.5f * x * (1.f + t);
}
__device__ __forceinline__ float gelu_bwd_f(float x) {
  float t = tanhf(0.79788456f * (x + 0.044715f * x * x * x));
  return 0.5f * x * (1.f - t * t) * (0.79788456f + 0.1070322243f * x * x) + 0.5f * (1.f + t);
}

// Block-cooperative GEMM over LDS buffers: Cf[M x N] (op)= A[M x K] @ B[K x N].
// MODE 0: store, MODE 1: subtract (RMW).
// BMODE 0: row-major B, 1: transposed (N-major) B, 2: fragment-packed B.
template<int MODE, int BMODE>
__device__ void blk_gemm(const __bf16* A, int lda, const __bf16* Bm, int ldb,
                         float* Cf, int ldc, int M, int N, int K) {
  int w = threadIdx.x >> 5;
  int ntiles = N >> 4;
  int T = (M >> 4) * ntiles;
  for (int t = w; t < T; t += 8) {
    int mt = t / ntiles, nt = t % ntiles;
    v8f acc = {};
    for (int k0 = 0; k0 < K; k0 += 32) {
      v16bf a = load_a_frag(A + mt * 16 * lda + k0, lda);
      v16bf b;
      if (BMODE == 0)      b = load_b_frag(Bm + k0 * ldb + nt * 16, ldb);
      else if (BMODE == 1) b = load_b_frag_T(Bm + (nt * 16) * ldb + k0, ldb);
      else                 b = load_b_frag_packed(Bm, k0 >> 5, nt, ntiles);
      acc = wmma_bf16(a, b, acc);
    }
    float* Ct = Cf + mt * 16 * ldc + nt * 16;
    int l = threadIdx.x & 31, n = l & 15, g = l >> 4;
    if (MODE == 0) {
#pragma unroll
      for (int r = 0; r < 8; ++r) Ct[(8 * g + r) * ldc + n] = acc[r];
    } else {
#pragma unroll
      for (int r = 0; r < 8; ++r) Ct[(8 * g + r) * ldc + n] -= acc[r];
    }
  }
}

// ------------------------------ utility kernels ------------------------------

__global__ void transpose_cvt_kernel(const float* __restrict__ Wsrc, __bf16* __restrict__ Wt,
                                     int Nrows, int Kcols) {
  size_t id = (size_t)blockIdx.x * 256 + threadIdx.x;
  size_t tot = (size_t)Nrows * Kcols;
  if (id >= tot) return;
  int n = (int)(id % Nrows);
  int k = (int)(id / Nrows);
  Wt[(size_t)k * Nrows + n] = (__bf16)Wsrc[(size_t)n * Kcols + k];
}

__global__ void cvt_scale_kernel(const float* __restrict__ X, __bf16* __restrict__ Y,
                                 float scale, int n) {
  int i = blockIdx.x * 256 + threadIdx.x;
  if (i < n) Y[i] = (__bf16)(X[i] * scale);
}

__global__ void add_kernel(const float* __restrict__ A, const float* __restrict__ Bv,
                           float* __restrict__ Y, int n) {
  int i = blockIdx.x * 256 + threadIdx.x;
  if (i < n) Y[i] = A[i] + Bv[i];
}

__global__ void final_gate_kernel(const float* __restrict__ resid, const float* __restrict__ z,
                                  const float* __restrict__ gate, float* __restrict__ out, int n) {
  int i = blockIdx.x * 256 + threadIdx.x;
  if (i < n) out[i] = resid[i] + tanhf(gate[i & (kD - 1)]) * z[i];
}

__global__ __launch_bounds__(256) void ln_to_bf16_kernel(const float* __restrict__ X,
                                                         const float* __restrict__ gw,
                                                         const float* __restrict__ gb,
                                                         __bf16* __restrict__ Y) {
  __shared__ float s1[256], s2[256];
  int row = blockIdx.x;
  const float* xr = X + (size_t)row * kD;
  float v[4], a = 0.f, q = 0.f;
#pragma unroll
  for (int i = 0; i < 4; ++i) {
    v[i] = xr[threadIdx.x + 256 * i];
    a += v[i]; q += v[i] * v[i];
  }
  s1[threadIdx.x] = a; s2[threadIdx.x] = q;
  __syncthreads();
  for (int off = 128; off; off >>= 1) {
    if (threadIdx.x < off) { s1[threadIdx.x] += s1[threadIdx.x + off]; s2[threadIdx.x] += s2[threadIdx.x + off]; }
    __syncthreads();
  }
  float mu = s1[0] * (1.f / kD);
  float var = s2[0] * (1.f / kD) - mu * mu;
  float rstd = rsqrtf(var + kEPS);
  __bf16* yr = Y + (size_t)row * kD;
#pragma unroll
  for (int i = 0; i < 4; ++i) {
    int c = threadIdx.x + 256 * i;
    yr[c] = (__bf16)((v[i] - mu) * rstd * gw[c] + gb[c]);
  }
}

// ------------------------------ tiled GEMM ----------------------------------
// Y[M x N] (f32) = A[M x K](bf16) @ Bt[K x N](bf16) (+ bias[N])
// grid: (ceil(N/64), M/64); block 256. A tile staged via async DMA; B tile
// staged into fragment-packed LDS layout (read as 2x ds_load_b128 per frag).
__global__ __launch_bounds__(256) void gemm_bf16_kernel(const __bf16* __restrict__ A,
                                                        const __bf16* __restrict__ Bt,
                                                        const float* __restrict__ bias,
                                                        float* __restrict__ Y,
                                                        int M, int N, int K) {
  __shared__ __align__(16) __bf16 sA[64][40];          // 80B rows (16B-aligned chunks)
  __shared__ __align__(16) __bf16 sBp[4 * 32 * 16];    // packed 32x64 B tile
  int n0 = blockIdx.x * 64, m0 = blockIdx.y * 64;
  bool full = (m0 + 64 <= M) && (n0 + 64 <= N);
  int w = threadIdx.x >> 5;
  int t0 = w, t1 = w + 8;
  v8f acc0 = {}, acc1 = {};
  for (int k0 = 0; k0 < K; k0 += 32) {
    __syncthreads();
    if (full) {
      {
        int r = threadIdx.x >> 2, ch = threadIdx.x & 3;   // A: 64 rows x 4 chunks (async DMA)
        async_copy16(&A[(size_t)(m0 + r) * K + k0 + ch * 8], &sA[r][ch * 8]);
      }
      {
        int r = threadIdx.x >> 3, ch = threadIdx.x & 7;   // B: coalesced load + packed scatter
        v8bf val = *(const v8bf*)&Bt[(size_t)(k0 + r) * N + n0 + ch * 8];
#pragma unroll
        for (int e = 0; e < 8; ++e) {
          int c = ch * 8 + e;
          sBp[bpack(r, c, 4)] = val[e];
        }
      }
      if (k0 + 32 < K) {
        __builtin_prefetch(&A[(size_t)(m0 + (threadIdx.x & 63)) * K + k0 + 32], 0, 1);
        __builtin_prefetch(&Bt[(size_t)(k0 + 32 + (threadIdx.x & 31)) * N + n0], 0, 1);
      }
      wait_async_lds();
    } else {
      for (int i = threadIdx.x; i < 64 * 32; i += 256) {
        int r = i >> 5, c = i & 31;
        int gr = m0 + r;
        sA[r][c] = (gr < M) ? A[(size_t)gr * K + k0 + c] : (__bf16)0.f;
      }
      for (int i = threadIdx.x; i < 32 * 64; i += 256) {
        int r = i >> 6, c = i & 63;
        int gc = n0 + c;
        sBp[bpack(r, c, 4)] = (gc < N) ? Bt[(size_t)(k0 + r) * N + gc] : (__bf16)0.f;
      }
    }
    __syncthreads();
    {
      int mt = t0 >> 2, nt = t0 & 3;
      v16bf a = load_a_frag(&sA[mt * 16][0], 40);
      v16bf b = load_b_frag_packed(sBp, 0, nt, 4);
      acc0 = wmma_bf16(a, b, acc0);
    }
    {
      int mt = t1 >> 2, nt = t1 & 3;
      v16bf a = load_a_frag(&sA[mt * 16][0], 40);
      v16bf b = load_b_frag_packed(sBp, 0, nt, 4);
      acc1 = wmma_bf16(a, b, acc1);
    }
  }
  int l = threadIdx.x & 31, n = l & 15, g = l >> 4;
  auto store_tile = [&](int t, v8f acc) {
    int mt = t >> 2, nt = t & 3;
#pragma unroll
    for (int r = 0; r < 8; ++r) {
      int gr = m0 + mt * 16 + 8 * g + r;
      int gc = n0 + nt * 16 + n;
      if (gr < M && gc < N)
        Y[(size_t)gr * N + gc] = acc[r] + (bias ? bias[gc] : 0.f);
    }
  };
  store_tile(t0, acc0);
  store_tile(t1, acc1);
}

// ------------------------------ flash attention ------------------------------
__global__ __launch_bounds__(32) void attn_kernel(const __bf16* __restrict__ Qb,
                                                  const __bf16* __restrict__ Kb,
                                                  const __bf16* __restrict__ Vb,
                                                  float* __restrict__ O) {
  const int ld = kD;
  int qt = blockIdx.x, h = blockIdx.y, bb = blockIdx.z;
  __shared__ __align__(16) float sS[16][36];
  __shared__ __align__(16) __bf16 sV[32][64];
  __shared__ float rmax[16], rsum[16], fac[16];
  int l = threadIdx.x, n = l & 15, g = l >> 4;
  if (l < 16) { rmax[l] = -1e30f; rsum[l] = 0.f; }
  const __bf16* Qp = Qb + ((size_t)(bb * kS + qt * 16)) * ld + h * kHD;
  v16bf aQ0 = load_a_frag(Qp, ld);
  v16bf aQ1 = load_a_frag(Qp + 32, ld);
  v8f o0 = {}, o1 = {}, o2 = {}, o3 = {};
  int qlast = qt * 16 + 15;
  __syncthreads();
  for (int kv0 = 0; kv0 <= qlast; kv0 += 32) {
    for (int i = l; i < 256; i += 32) {      // async V-tile DMA, overlaps QK^T
      int r = i >> 3, ch = i & 7;
      async_copy16(&Vb[((size_t)(bb * kS + kv0 + r)) * ld + h * kHD + ch * 8], &sV[r][ch * 8]);
    }
#pragma unroll
    for (int ntile = 0; ntile < 2; ++ntile) {
      const __bf16* Kp = Kb + ((size_t)(bb * kS + kv0 + ntile * 16)) * ld + h * kHD;
      v8f s = {};
      s = wmma_bf16(aQ0, load_b_frag_T(Kp, ld), s);
      s = wmma_bf16(aQ1, load_b_frag_T(Kp + 32, ld), s);
      store_c_frag(&sS[0][ntile * 16], 36, s);
    }
    __syncthreads();
    if (l < 16) {
      int q = qt * 16 + l;
      int jmax = q - kv0; if (jmax > 31) jmax = 31;
      float mx = rmax[l];
      float sc, ts = 0.f;
      if (jmax < 0) {
        sc = 1.f;
        for (int j = 0; j < 32; ++j) sS[l][j] = 0.f;
      } else {
        float tm = -1e30f;
        for (int j = 0; j <= jmax; ++j) tm = fmaxf(tm, sS[l][j]);
        float nm = fmaxf(mx, tm);
        sc = __expf(mx - nm);
        for (int j = 0; j < 32; ++j) {
          float p = (j <= jmax) ? __expf(sS[l][j] - nm) : 0.f;
          sS[l][j] = p; ts += p;
        }
        rmax[l] = nm;
      }
      rsum[l] = rsum[l] * sc + ts;
      fac[l] = sc;
    }
    __syncthreads();
    v16bf aP = load_a_frag_f32(&sS[0][0], 36);
    float f[8];
#pragma unroll
    for (int r = 0; r < 8; ++r) f[r] = fac[8 * g + r];
#pragma unroll
    for (int r = 0; r < 8; ++r) { o0[r] *= f[r]; o1[r] *= f[r]; o2[r] *= f[r]; o3[r] *= f[r]; }
    wait_async_lds();
    __syncthreads();
    o0 = wmma_bf16(aP, load_b_frag(&sV[0][0], 64), o0);
    o1 = wmma_bf16(aP, load_b_frag(&sV[0][16], 64), o1);
    o2 = wmma_bf16(aP, load_b_frag(&sV[0][32], 64), o2);
    o3 = wmma_bf16(aP, load_b_frag(&sV[0][48], 64), o3);
    __syncthreads();
  }
  float inv[8];
#pragma unroll
  for (int r = 0; r < 8; ++r) inv[r] = 1.f / rsum[8 * g + r];
  float* Op = O + ((size_t)(bb * kS + qt * 16)) * ld + h * kHD;
#pragma unroll
  for (int r = 0; r < 8; ++r) {
    int row = 8 * g + r;
    Op[(size_t)row * ld + 0 + n]  = o0[r] * inv[r];
    Op[(size_t)row * ld + 16 + n] = o1[r] * inv[r];
    Op[(size_t)row * ld + 32 + n] = o2[r] * inv[r];
    Op[(size_t)row * ld + 48 + n] = o3[r] * inv[r];
  }
}

// --------------------- TTT input normalization (per head) --------------------
__global__ __launch_bounds__(256) void ttt_prenorm_kernel(const float* __restrict__ XQf,
                                                          const float* __restrict__ XKf,
                                                          const float* __restrict__ XVf,
                                                          const float* __restrict__ nw,
                                                          const float* __restrict__ nb,
                                                          __bf16* __restrict__ XQn,
                                                          __bf16* __restrict__ XKn,
                                                          __bf16* __restrict__ XVn) {
  int w = threadIdx.x >> 5, l = threadIdx.x & 31;
  int blk = blockIdx.x;
  int half = blk & 1, tok = blk >> 1;
  int h = half * 8 + w;
  size_t base = ((size_t)tok * kH + h) * kHD;
  int d0 = l * 2;
  float q0 = XQf[base + d0], q1 = XQf[base + d0 + 1];
  float nq = fmaxf(sqrtf(wave_sum(q0 * q0 + q1 * q1)), 1e-12f);
  XQn[base + d0]     = (__bf16)(q0 / nq);
  XQn[base + d0 + 1] = (__bf16)(q1 / nq);
  float k0 = XKf[base + d0], k1 = XKf[base + d0 + 1];
  float nk = fmaxf(sqrtf(wave_sum(k0 * k0 + k1 * k1)), 1e-12f);
  float ka = k0 / nk, kb2 = k1 / nk;
  XKn[base + d0]     = (__bf16)ka;
  XKn[base + d0 + 1] = (__bf16)kb2;
  float v0 = XVf[base + d0], v1 = XVf[base + d0 + 1];
  float mu = wave_sum(v0 + v1) * (1.f / kHD);
  float e0 = v0 - mu, e1 = v1 - mu;
  float var = wave_sum(e0 * e0 + e1 * e1) * (1.f / kHD);
  float rstd = rsqrtf(var + kEPS);
  float g0 = nw[h * kHD + d0], g1 = nw[h * kHD + d0 + 1];
  float b0 = nb[h * kHD + d0], b1 = nb[h * kHD + d0 + 1];
  XVn[base + d0]     = (__bf16)(g0 * (e0 * rstd) + b0 + ka);
  XVn[base + d0 + 1] = (__bf16)(g1 * (e1 * rstd) + b1 + kb2);
}

__global__ void eta_kernel(const float* __restrict__ logits, const float* __restrict__ lr_b,
                           float* __restrict__ eta) {
  int id = blockIdx.x * 256 + threadIdx.x;
  if (id >= kB * kH * kS) return;
  int s = id % kS, bh = id / kS;
  int h = bh % kH, bb = bh / kH;
  float lg = logits[((size_t)(bb * kS + s)) * kH + h] + lr_b[h];
  eta[id] = (1.f / (1.f + __expf(-lg))) * (1.f / kHD);
}

// ------------------------------ TTT scan kernel ------------------------------
// One block (256 thr / 8 waves) per (h, b). W1 and gZ1 live in fragment-packed
// layout (B-only operands); gZ2 keeps natural (A-side) + packed (B-side) copies.
__global__ __launch_bounds__(256) void ttt_scan_kernel(const __bf16* __restrict__ XQn,
                                                       const __bf16* __restrict__ XKn,
                                                       const __bf16* __restrict__ XVn,
                                                       const float* __restrict__ eta,
                                                       const float* __restrict__ W1g,
                                                       const float* __restrict__ b1g,
                                                       const float* __restrict__ W2g,
                                                       const float* __restrict__ b2g,
                                                       const float* __restrict__ nw,
                                                       const float* __restrict__ nb,
                                                       float* __restrict__ ttt_out) {
  extern __shared__ char smem[];
  __bf16* sW1  = (__bf16*)smem;          // [64][256] fragment-PACKED (B operand only)
  __bf16* sW2  = sW1 + 64 * 256;         // [256][64] natural (B and B^T uses)
  __bf16* sZ1  = sW2 + 256 * 64;         // [64][256] natural; later X2b; later T/T2
  __bf16* sX2  = sZ1 + 64 * 256;         // [64][256] natural (A and B^T uses)
  __bf16* sGZ1 = sX2 + 64 * 256;         // [64][256] fragment-PACKED (B operand only)
  __bf16* sXQ  = sGZ1 + 64 * 256;        // [64][64]
  __bf16* sXK  = sXQ + 64 * 64;
  __bf16* sXV  = sXK + 64 * 64;
  __bf16* sGZ2 = sXV + 64 * 64;          // natural (A use, b2 update)
  __bf16* sM   = sGZ2 + 64 * 64;
  __bf16* sGZ2p= sM + 64 * 64;           // [64][64] fragment-PACKED mirror of gZ2
  float*  sF   = (float*)(sGZ2p + 64 * 64); // [64][256] f32 GEMM scratch
  float*  sb1  = sF + 64 * 256;          // [256]
  float*  sb2  = sb1 + 256;              // [64]
  float*  sEta = sb2 + 64;               // [64]
  float*  sgW  = sEta + 64;              // [64]
  float*  sgB  = sgW + 64;               // [64]

  int h = blockIdx.x, bb = blockIdx.y;
  int tid = threadIdx.x;

  for (int i = tid; i < 64 * 256; i += 256) {
    int k = i >> 8, n = i & 255;
    sW1[bpack(k, n, 16)] = (__bf16)W1g[(size_t)h * 64 * 256 + i];
  }
  for (int i = tid; i < 256 * 64; i += 256) sW2[i] = (__bf16)W2g[(size_t)h * 256 * 64 + i];
  if (tid < 256) sb1[tid] = b1g[h * 256 + tid];
  if (tid < 64) {
    sb2[tid] = b2g[h * 64 + tid];
    sgW[tid] = nw[h * 64 + tid];
    sgB[tid] = nb[h * 64 + tid];
  }

  for (int chunk = 0; chunk < kNC; ++chunk) {
    __syncthreads();
    size_t tokbase = (size_t)bb * kS + chunk * 64;
    {
      size_t gb = tokbase * (kH * kHD) + (size_t)h * kHD;
      for (int i = tid; i < 512; i += 256) {
        int r = i >> 3, ch = i & 7;
        size_t go = gb + (size_t)r * (kH * kHD) + ch * 8;
        int lo = r * 64 + ch * 8;
        async_copy16(&XQn[go], &sXQ[lo]);
        async_copy16(&XKn[go], &sXK[lo]);
        async_copy16(&XVn[go], &sXV[lo]);
      }
    }
    if (tid < 64) sEta[tid] = eta[((size_t)bb * kH + h) * kS + chunk * 64 + tid];
    wait_async_lds();
    __syncthreads();

    // (1) Z1 = xk @ W1 + b1 ; X2 = gelu(Z1)
    blk_gemm<0, 2>(sXK, 64, sW1, 0, sF, 256, 64, 256, 64);
    __syncthreads();
    for (int i = tid; i < 64 * 256; i += 256) {
      float z = sF[i] + sb1[i & 255];
      sZ1[i] = (__bf16)z;
      sX2[i] = (__bf16)gelu_f(z);
    }
    __syncthreads();

    // (2) Z2 = X2 @ W2 + b2 ; gZ2 = ln_l2_bwd(Z2, xv - xk, gW, gB)
    blk_gemm<0, 0>(sX2, 256, sW2, 64, sF, 64, 64, 64, 256);
    __syncthreads();
    {
      int w = tid >> 5, l = tid & 31;
      for (int r = w * 8; r < w * 8 + 8; ++r) {
        float z0 = sF[r * 64 + l] + sb2[l];
        float z1 = sF[r * 64 + l + 32] + sb2[l + 32];
        float mu = wave_sum(z0 + z1) * (1.f / 64.f);
        float e0 = z0 - mu, e1 = z1 - mu;
        float var = wave_sum(e0 * e0 + e1 * e1) * (1.f / 64.f);
        float rstd = rsqrtf(var + kEPS);
        float xh0 = e0 * rstd, xh1 = e1 * rstd;
        float g0 = sgW[l], g1 = sgW[l + 32];
        float t0 = (float)sXV[r * 64 + l] - (float)sXK[r * 64 + l];
        float t1 = (float)sXV[r * 64 + l + 32] - (float)sXK[r * 64 + l + 32];
        float gh0 = (g0 * xh0 + sgB[l] - t0) * g0;
        float gh1 = (g1 * xh1 + sgB[l + 32] - t1) * g1;
        float sgh  = wave_sum(gh0 + gh1);
        float sghx = wave_sum(gh0 * xh0 + gh1 * xh1);
        float gz0 = (64.f * gh0 - sgh - xh0 * sghx) * rstd * (1.f / 64.f);
        float gz1 = (64.f * gh1 - sgh - xh1 * sghx) * rstd * (1.f / 64.f);
        sGZ2[r * 64 + l]       = (__bf16)gz0;
        sGZ2[r * 64 + l + 32]  = (__bf16)gz1;
        sGZ2p[bpack(r, l, 4)]      = (__bf16)gz0;
        sGZ2p[bpack(r, l + 32, 4)] = (__bf16)gz1;
      }
    }
    __syncthreads();

    // (3) gZ1 = (gZ2 @ W2^T) * gelu_bwd(Z1)   -> packed layout
    blk_gemm<0, 1>(sGZ2, 64, sW2, 64, sF, 256, 64, 256, 64);
    __syncthreads();
    for (int i = tid; i < 64 * 256; i += 256) {
      int j = i >> 8, c = i & 255;
      sGZ1[bpack(j, c, 16)] = (__bf16)(sF[i] * gelu_bwd_f((float)sZ1[i]));
    }
    __syncthreads();

    // (4) M1[i][j] = (j<=i) ? eta[j]*(A1raw[i][j] + 1) : 0 ;  A1raw = xq@xk^T
    blk_gemm<0, 1>(sXQ, 64, sXK, 64, sF, 64, 64, 64, 64);
    __syncthreads();
    for (int i = tid; i < 64 * 64; i += 256) {
      int r = i >> 6, c = i & 63;
      sM[i] = (__bf16)((c <= r) ? sEta[c] * (sF[i] + 1.f) : 0.f);
    }
    __syncthreads();

    // (5) Z1b = xq@W1 + b1 - M1@gZ1 ; X2b = gelu(Z1b)
    blk_gemm<0, 2>(sXQ, 64, sW1, 0, sF, 256, 64, 256, 64);
    __syncthreads();
    blk_gemm<1, 2>(sM, 64, sGZ1, 0, sF, 256, 64, 256, 64);
    __syncthreads();
    __bf16* sX2b = sZ1;
    for (int i = tid; i < 64 * 256; i += 256)
      sX2b[i] = (__bf16)gelu_f(sF[i] + sb1[i & 255]);
    __syncthreads();

    // (6) M2 from A2raw = X2b @ X2^T  (K = 256)
    blk_gemm<0, 1>(sX2b, 256, sX2, 256, sF, 64, 64, 64, 256);
    __syncthreads();
    for (int i = tid; i < 64 * 64; i += 256) {
      int r = i >> 6, c = i & 63;
      sM[i] = (__bf16)((c <= r) ? sEta[c] * (sF[i] + 1.f) : 0.f);
    }
    __syncthreads();

    // (7) Z2b = X2b@W2 + b2 - M2@gZ2 ; out = xq + LN(Z2b; gW,gB)
    blk_gemm<0, 0>(sX2b, 256, sW2, 64, sF, 64, 64, 64, 256);
    __syncthreads();
    blk_gemm<1, 2>(sM, 64, sGZ2p, 0, sF, 64, 64, 64, 64);
    __syncthreads();
    {
      int w = tid >> 5, l = tid & 31;
      for (int r = w * 8; r < w * 8 + 8; ++r) {
        float z0 = sF[r * 64 + l] + sb2[l];
        float z1 = sF[r * 64 + l + 32] + sb2[l + 32];
        float mu = wave_sum(z0 + z1) * (1.f / 64.f);
        float e0 = z0 - mu, e1 = z1 - mu;
        float var = wave_sum(e0 * e0 + e1 * e1) * (1.f / 64.f);
        float rstd = rsqrtf(var + kEPS);
        size_t o = (tokbase + r) * (size_t)kD + h * kHD;
        ttt_out[o + l]      = e0 * rstd * sgW[l]      + sgB[l]      + (float)sXQ[r * 64 + l];
        ttt_out[o + l + 32] = e1 * rstd * sgW[l + 32] + sgB[l + 32] + (float)sXQ[r * 64 + l + 32];
      }
    }
    __syncthreads();

    // (8) W1 -= (eta⊙xk)^T @ gZ1 ; b1 -= sum_j eta[j]*gZ1[j]
    __bf16* sT = sZ1;
    for (int i = tid; i < 64 * 64; i += 256) {
      int d = i >> 6, j = i & 63;
      sT[i] = (__bf16)(sEta[j] * (float)sXK[j * 64 + d]);
    }
    __syncthreads();
    blk_gemm<0, 2>(sT, 64, sGZ1, 0, sF, 256, 64, 256, 64);
    __syncthreads();
    for (int i = tid; i < 64 * 256; i += 256) {
      int k = i >> 8, n = i & 255;
      int p = bpack(k, n, 16);
      sW1[p] = (__bf16)((float)sW1[p] - sF[i]);
    }
    if (tid < 256) {
      float acc = 0.f;
      for (int j = 0; j < 64; ++j) acc += sEta[j] * (float)sGZ1[bpack(j, tid, 16)];
      sb1[tid] -= acc;
    }
    __syncthreads();

    // (9) W2 -= (eta⊙X2)^T @ gZ2 ; b2 -= sum_j eta[j]*gZ2[j]
    for (int i = tid; i < 256 * 64; i += 256) {
      int p = i >> 6, j = i & 63;
      sT[i] = (__bf16)(sEta[j] * (float)sX2[j * 256 + p]);
    }
    __syncthreads();
    blk_gemm<0, 2>(sT, 64, sGZ2p, 0, sF, 64, 256, 64, 64);
    __syncthreads();
    for (int i = tid; i < 256 * 64; i += 256)
      sW2[i] = (__bf16)((float)sW2[i] - sF[i]);
    if (tid < 64) {
      float acc = 0.f;
      for (int j = 0; j < 64; ++j) acc += sEta[j] * (float)sGZ2[j * 64 + tid];
      sb2[tid] -= acc;
    }
    __syncthreads();
  }
}

// ------------------------------ orchestration --------------------------------

extern "C" void kernel_launch(void* const* d_in, const int* in_sizes, int n_in,
                              void* d_out, int out_size, void* d_ws, size_t ws_size,
                              hipStream_t stream) {
  (void)in_sizes; (void)n_in; (void)out_size; (void)ws_size;
  const float* x       = (const float*)d_in[0];
  const float* ln1_w   = (const float*)d_in[1];
  const float* ln1_b   = (const float*)d_in[2];
  const float* attn_wq = (const float*)d_in[3];
  const float* attn_wk = (const float*)d_in[4];
  const float* attn_wv = (const float*)d_in[5];
  const float* attn_wo = (const float*)d_in[6];
  const float* wq_w    = (const float*)d_in[7];
  const float* wq_b    = (const float*)d_in[8];
  const float* wk_w    = (const float*)d_in[9];
  const float* wk_b    = (const float*)d_in[10];
  const float* wv_w    = (const float*)d_in[11];
  const float* wv_b    = (const float*)d_in[12];
  const float* wo_w    = (const float*)d_in[13];
  const float* wo_b    = (const float*)d_in[14];
  const float* W1g     = (const float*)d_in[15];
  const float* b1g     = (const float*)d_in[16];
  const float* W2g     = (const float*)d_in[17];
  const float* b2g     = (const float*)d_in[18];
  const float* ttt_nw  = (const float*)d_in[19];
  const float* ttt_nb  = (const float*)d_in[20];
  const float* lr_w    = (const float*)d_in[21];
  const float* lr_b    = (const float*)d_in[22];
  const float* pn_w    = (const float*)d_in[23];
  const float* pn_b    = (const float*)d_in[24];
  const float* gate_a  = (const float*)d_in[25];
  float* out = (float*)d_out;

  char* ws = (char*)d_ws;
  auto carve = [&](size_t bytes) -> void* {
    void* p = (void*)ws;
    ws += (bytes + 255) & ~(size_t)255;
    return p;
  };

  const size_t TOK = (size_t)kB * kS;
  const size_t TD  = TOK * kD;

  __bf16* wT_aq = (__bf16*)carve((size_t)kD * kD * 2);
  __bf16* wT_ak = (__bf16*)carve((size_t)kD * kD * 2);
  __bf16* wT_av = (__bf16*)carve((size_t)kD * kD * 2);
  __bf16* wT_ao = (__bf16*)carve((size_t)kD * kD * 2);
  __bf16* wT_q  = (__bf16*)carve((size_t)kD * kD * 2);
  __bf16* wT_k  = (__bf16*)carve((size_t)kD * kD * 2);
  __bf16* wT_v  = (__bf16*)carve((size_t)kD * kD * 2);
  __bf16* wT_o  = (__bf16*)carve((size_t)kD * kD * 2);
  __bf16* wT_lr = (__bf16*)carve((size_t)kD * kH * 2);
  __bf16* h_b   = (__bf16*)carve(TD * 2);
  float*  qf    = (float*)carve(TD * 4);
  float*  kf    = (float*)carve(TD * 4);
  float*  vf    = (float*)carve(TD * 4);
  __bf16* qb    = (__bf16*)carve(TD * 2);
  __bf16* kb    = (__bf16*)carve(TD * 2);
  __bf16* vb    = (__bf16*)carve(TD * 2);
  float*  of    = (float*)carve(TD * 4);
  __bf16* ob    = (__bf16*)carve(TD * 2);
  float*  xa    = (float*)carve(TD * 4);
  __bf16* xab   = (__bf16*)carve(TD * 2);
  float*  logit = (float*)carve(TOK * kH * 4);
  float*  eta   = (float*)carve((size_t)kB * kH * kS * 4);
  float*  attno  = qf;
  float*  XQf = qf;  float* XKf = kf;  float* XVf = vf;
  __bf16* XQn = qb;  __bf16* XKn = kb;  __bf16* XVn = vb;
  float*  tttout = of;
  __bf16* tb     = ob;
  float*  zf     = vf;

  dim3 blk256(256), blk32(32);
  const int tg   = (kD * kD + 255) / 256;
  const int cgr  = (int)((TD + 255) / 256);
  dim3 ggrid(kD / 64, (unsigned)(TOK / 64));

  transpose_cvt_kernel<<<tg, blk256, 0, stream>>>(attn_wq, wT_aq, kD, kD);
  transpose_cvt_kernel<<<tg, blk256, 0, stream>>>(attn_wk, wT_ak, kD, kD);
  transpose_cvt_kernel<<<tg, blk256, 0, stream>>>(attn_wv, wT_av, kD, kD);
  transpose_cvt_kernel<<<tg, blk256, 0, stream>>>(attn_wo, wT_ao, kD, kD);
  transpose_cvt_kernel<<<tg, blk256, 0, stream>>>(wq_w, wT_q, kD, kD);
  transpose_cvt_kernel<<<tg, blk256, 0, stream>>>(wk_w, wT_k, kD, kD);
  transpose_cvt_kernel<<<tg, blk256, 0, stream>>>(wv_w, wT_v, kD, kD);
  transpose_cvt_kernel<<<tg, blk256, 0, stream>>>(wo_w, wT_o, kD, kD);
  transpose_cvt_kernel<<<(kD * kH + 255) / 256, blk256, 0, stream>>>(lr_w, wT_lr, kH, kD);

  ln_to_bf16_kernel<<<(unsigned)TOK, blk256, 0, stream>>>(x, ln1_w, ln1_b, h_b);

  gemm_bf16_kernel<<<ggrid, blk256, 0, stream>>>(h_b, wT_aq, nullptr, qf, (int)TOK, kD, kD);
  gemm_bf16_kernel<<<ggrid, blk256, 0, stream>>>(h_b, wT_ak, nullptr, kf, (int)TOK, kD, kD);
  gemm_bf16_kernel<<<ggrid, blk256, 0, stream>>>(h_b, wT_av, nullptr, vf, (int)TOK, kD, kD);
  cvt_scale_kernel<<<cgr, blk256, 0, stream>>>(qf, qb, 0.125f, (int)TD);
  cvt_scale_kernel<<<cgr, blk256, 0, stream>>>(kf, kb, 1.f, (int)TD);
  cvt_scale_kernel<<<cgr, blk256, 0, stream>>>(vf, vb, 1.f, (int)TD);

  dim3 agrid(kS / 16, kH, kB);
  attn_kernel<<<agrid, blk32, 0, stream>>>(qb, kb, vb, of);

  cvt_scale_kernel<<<cgr, blk256, 0, stream>>>(of, ob, 1.f, (int)TD);
  gemm_bf16_kernel<<<ggrid, blk256, 0, stream>>>(ob, wT_ao, nullptr, attno, (int)TOK, kD, kD);
  add_kernel<<<cgr, blk256, 0, stream>>>(x, attno, xa, (int)TD);
  cvt_scale_kernel<<<cgr, blk256, 0, stream>>>(xa, xab, 1.f, (int)TD);

  gemm_bf16_kernel<<<ggrid, blk256, 0, stream>>>(xab, wT_q, wq_b, XQf, (int)TOK, kD, kD);
  gemm_bf16_kernel<<<ggrid, blk256, 0, stream>>>(xab, wT_k, wk_b, XKf, (int)TOK, kD, kD);
  gemm_bf16_kernel<<<ggrid, blk256, 0, stream>>>(xab, wT_v, wv_b, XVf, (int)TOK, kD, kD);

  dim3 lgrid(1, (unsigned)(TOK / 64));
  gemm_bf16_kernel<<<lgrid, blk256, 0, stream>>>(xab, wT_lr, nullptr, logit, (int)TOK, kH, kD);
  eta_kernel<<<(kB * kH * kS) / 256, blk256, 0, stream>>>(logit, lr_b, eta);

  ttt_prenorm_kernel<<<(unsigned)(TOK * 2), blk256, 0, stream>>>(XQf, XKf, XVf, ttt_nw, ttt_nb,
                                                                 XQn, XKn, XVn);

  constexpr size_t SCAN_LDS =
      (size_t)(5 * 64 * 256 + 6 * 64 * 64) * 2 +                      // bf16 buffers
      (size_t)(64 * 256 + 256 + 64 + 64 + 64 + 64) * 4;               // f32 buffers
  dim3 sgrid(kH, kB);
  ttt_scan_kernel<<<sgrid, blk256, SCAN_LDS, stream>>>(XQn, XKn, XVn, eta, W1g, b1g, W2g, b2g,
                                                       ttt_nw, ttt_nb, tttout);

  ln_to_bf16_kernel<<<(unsigned)TOK, blk256, 0, stream>>>(tttout, pn_w, pn_b, tb);
  gemm_bf16_kernel<<<ggrid, blk256, 0, stream>>>(tb, wT_o, wo_b, zf, (int)TOK, kD, kD);
  final_gate_kernel<<<cgr, blk256, 0, stream>>>(xa, zf, gate_a, out, (int)TD);
}
